// GAT_48524540510808
// MI455X (gfx1250) — compile-verified, matrix operand order checked by reading, not run on hardware
//
#include <hip/hip_runtime.h>
#include <hip/hip_bf16.h>
#include <math.h>

// ---------------------------------------------------------------------------
// GAT 2-layer forward, MI455X (gfx1250, wave32).
// GEMMs: v_wmma_f32_16x16x32_bf16, one wave per 16-row strip with NT column
// tiles; B pre-packed into per-lane WMMA fragments (bf16) by a tiny kernel.
// Edge softmax/aggregation via streaming kernels + native fp32 atomics.
// ---------------------------------------------------------------------------

#define GAT_N      50000
#define GAT_E      800000
#define GAT_INC    512
#define GAT_MIDC   8
#define GAT_HEADS  8
#define GAT_OUTC   16
#define GAT_NEG    0.28f
#define GAT_EPS    1e-16f

typedef __attribute__((ext_vector_type(16))) __bf16 v16bf;
typedef __attribute__((ext_vector_type(8)))  float  v8f;

// --- monotone uint key for float atomic max ---------------------------------
__device__ __forceinline__ unsigned fkey(float f) {
    unsigned u = __float_as_uint(f);
    return (u & 0x80000000u) ? ~u : (u | 0x80000000u);
}
__device__ __forceinline__ float funkey(unsigned k) {
    unsigned u = (k & 0x80000000u) ? (k & 0x7FFFFFFFu) : ~k;
    return __uint_as_float(u);
}
#define FKEY_NEG_INF 0x007FFFFFu   // fkey(-inf)

// --- fills ------------------------------------------------------------------
__global__ void fill_u32_kernel(unsigned* __restrict__ p, unsigned v, int n) {
    int i = blockIdx.x * blockDim.x + threadIdx.x;
    if (i < n) p[i] = v;
}
__global__ void fill_f32_kernel(float* __restrict__ p, float v, int n) {
    int i = blockIdx.x * blockDim.x + threadIdx.x;
    if (i < n) p[i] = v;
}

// --- repack W[K,Nc] (fp32, row-major) into per-lane bf16 WMMA B fragments ---
// Fragment (tn, ks): 32 lanes x 16 bf16, contiguous. Bp index:
//   Bp[(tn*ksteps + ks)*32 + lane]
// ISA 16-bit B layout (mirrors A): lanes 0-15 K={0..7,16..23}, lanes 16-31
// K={8..15,24..31}; column N = lane&15; pairs packed per dword.
__global__ void repack_B_kernel(const float* __restrict__ W,
                                v16bf* __restrict__ Bp,
                                int K, int Nc) {
    const int ksteps = K >> 5;
    const int tilesN = Nc >> 4;
    const int total  = tilesN * ksteps * 32;
    int tid = blockIdx.x * blockDim.x + threadIdx.x;
    if (tid >= total) return;
    const int lane  = tid & 31;
    const int ks    = (tid >> 5) % ksteps;
    const int tn    = (tid >> 5) / ksteps;
    const int col   = tn * 16 + (lane & 15);
    const int khalf = lane >> 4;
    const int k0    = ks * 32;
    v16bf b;
#pragma unroll
    for (int i = 0; i < 8; ++i) {
        const int kk = (i < 4) ? (8 * khalf + 2 * i)
                               : (16 + 8 * khalf + 2 * (i - 4));
        b[2 * i]     = (__bf16)W[(size_t)(k0 + kk) * Nc + col];
        b[2 * i + 1] = (__bf16)W[(size_t)(k0 + kk + 1) * Nc + col];
    }
    Bp[tid] = b;
}

// --- bf16 WMMA GEMM: C[M, NT*16] = A[M,K] (fp32) * Bp (packed bf16) ---------
// One wave per 16-row strip; NT accumulators reuse one A fragment per K-step.
template <int NT>
__global__ void gemm_bf16_wmma_kernel(const float* __restrict__ A,
                                      const v16bf* __restrict__ Bp,
                                      float* __restrict__ C,
                                      int M, int K, int lda, int ldc) {
    const int wave = (blockIdx.x * blockDim.x + threadIdx.x) >> 5;
    const int lane = threadIdx.x & 31;
    const int tilesM = M >> 4;
    if (wave >= tilesM) return;           // wave-uniform: EXEC stays full
    const int mrow  = lane & 15;
    const int khalf = lane >> 4;
    const int ksteps = K >> 5;

    const float* arow = A + (size_t)(wave * 16 + mrow) * lda;

    v8f acc[NT];
#pragma unroll
    for (int t = 0; t < NT; ++t) acc[t] = (v8f){};

    for (int ks = 0; ks < ksteps; ++ks) {
        const int k0 = ks * 32;
        // A K-pattern per half-wave = two contiguous 8-float segments.
        const float4* s0 = reinterpret_cast<const float4*>(arow + k0 + 8 * khalf);
        const float4* s1 = reinterpret_cast<const float4*>(arow + k0 + 16 + 8 * khalf);
        float4 a0 = s0[0], a1 = s0[1];
        float4 a2 = s1[0], a3 = s1[1];
        v16bf a;
        a[0]  = (__bf16)a0.x; a[1]  = (__bf16)a0.y; a[2]  = (__bf16)a0.z; a[3]  = (__bf16)a0.w;
        a[4]  = (__bf16)a1.x; a[5]  = (__bf16)a1.y; a[6]  = (__bf16)a1.z; a[7]  = (__bf16)a1.w;
        a[8]  = (__bf16)a2.x; a[9]  = (__bf16)a2.y; a[10] = (__bf16)a2.z; a[11] = (__bf16)a2.w;
        a[12] = (__bf16)a3.x; a[13] = (__bf16)a3.y; a[14] = (__bf16)a3.z; a[15] = (__bf16)a3.w;
#pragma unroll
        for (int t = 0; t < NT; ++t) {
            v16bf b = Bp[((size_t)t * ksteps + ks) * 32 + lane];
            acc[t] = __builtin_amdgcn_wmma_f32_16x16x32_bf16(
                         false, a, false, b, (short)0, acc[t], false, false);
        }
    }
    // C/D layout: VGPR r holds M = r + 8*khalf, N = lane&15
#pragma unroll
    for (int t = 0; t < NT; ++t) {
        float* crow = C + (size_t)(wave * 16 + 8 * khalf) * ldc + t * 16 + mrow;
#pragma unroll
        for (int r = 0; r < 8; ++r)
            crow[(size_t)r * ldc] = acc[t][r];
    }
}

// --- per-node attention coefficients: as/ad[n,h] = <h[n,h,:], att[h,:]> -----
__global__ void node_alpha_kernel(const float* __restrict__ h,
                                  const float* __restrict__ a_src,
                                  const float* __restrict__ a_dst,
                                  float* __restrict__ as_,
                                  float* __restrict__ ad_,
                                  int n, int H, int C) {
    int nd = blockIdx.x * blockDim.x + threadIdx.x;
    if (nd >= n) return;
    const float* row = h + (size_t)nd * H * C;
    for (int hh = 0; hh < H; ++hh) {
        float s = 0.f, d = 0.f;
        for (int c = 0; c < C; ++c) {
            float v = row[hh * C + c];
            s += v * a_src[hh * C + c];
            d += v * a_dst[hh * C + c];
        }
        as_[(size_t)nd * H + hh] = s;
        ad_[(size_t)nd * H + hh] = d;
    }
}

// --- edge logits (leaky-relu) + segment max over dst ------------------------
__global__ void edge_logit_max_kernel(const long long* __restrict__ adj,
                                      const float* __restrict__ as_,
                                      const float* __restrict__ ad_,
                                      float* __restrict__ eval,
                                      unsigned* __restrict__ emaxk,
                                      int E, int H) {
    int tid = blockIdx.x * blockDim.x + threadIdx.x;
    if (tid >= E * H) return;
    int e  = tid / H;
    int hh = tid % H;
    int s  = (int)adj[e];
    int d  = (int)adj[E + e];
    float v = as_[(size_t)s * H + hh] + ad_[(size_t)d * H + hh];
    v = (v >= 0.f) ? v : GAT_NEG * v;
    eval[tid] = v;
    atomicMax(&emaxk[(size_t)d * H + hh], fkey(v));
}

// --- exp(e - max) + segment sum over dst ------------------------------------
__global__ void edge_exp_sum_kernel(const long long* __restrict__ adj,
                                    const float* __restrict__ eval,
                                    const unsigned* __restrict__ emaxk,
                                    float* __restrict__ eexp,
                                    float* __restrict__ denom,
                                    int E, int H) {
    int tid = blockIdx.x * blockDim.x + threadIdx.x;
    if (tid >= E * H) return;
    int e  = tid / H;
    int hh = tid % H;
    int d  = (int)adj[E + e];
    float mx = funkey(emaxk[(size_t)d * H + hh]);
    float ex = __expf(eval[tid] - mx);
    eexp[tid] = ex;
    atomicAdd(&denom[(size_t)d * H + hh], ex);
}

// --- normalized message scatter: agg[dst] += alpha * h[src] -----------------
// (H*C)/4 threads per edge, float4 gather, 4 scalar fp32 global atomics.
__global__ void edge_message_kernel(const long long* __restrict__ adj,
                                    const float* __restrict__ eexp,
                                    const float* __restrict__ denom,
                                    const float* __restrict__ h,
                                    float* __restrict__ agg,
                                    int E, int H, int C) {
    const int tpe = (H * C) >> 2;                // threads per edge
    long long tid = (long long)blockIdx.x * blockDim.x + threadIdx.x;
    long long e   = tid / tpe;
    int       j   = (int)(tid % tpe);
    if (e >= E) return;
    int s  = (int)adj[e];
    int d  = (int)adj[E + e];
    int hh = (4 * j) / C;
    float alpha = eexp[(size_t)e * H + hh] /
                  (denom[(size_t)d * H + hh] + GAT_EPS);
    const float4 hv = *reinterpret_cast<const float4*>(
        h + (size_t)s * H * C + 4 * j);
    float* out = agg + (size_t)d * H * C + 4 * j;
    atomicAdd(out + 0, alpha * hv.x);
    atomicAdd(out + 1, alpha * hv.y);
    atomicAdd(out + 2, alpha * hv.z);
    atomicAdd(out + 3, alpha * hv.w);
}

// --- bias + ELU -------------------------------------------------------------
__global__ void elu_bias_kernel(const float* __restrict__ agg,
                                const float* __restrict__ bias,
                                float* __restrict__ out, int n, int F) {
    int i = blockIdx.x * blockDim.x + threadIdx.x;
    if (i >= n * F) return;
    float v = agg[i] + bias[i % F];
    out[i] = (v > 0.f) ? v : expm1f(v);
}

// --- head-mean + bias + log_softmax -----------------------------------------
__global__ void final_kernel(const float* __restrict__ agg,
                             const float* __restrict__ b2,
                             float* __restrict__ out, int n) {
    int nd = blockIdx.x * blockDim.x + threadIdx.x;
    if (nd >= n) return;
    float z[GAT_OUTC];
    const float* row = agg + (size_t)nd * GAT_HEADS * GAT_OUTC;
#pragma unroll
    for (int c = 0; c < GAT_OUTC; ++c) {
        float s = 0.f;
#pragma unroll
        for (int hh = 0; hh < GAT_HEADS; ++hh)
            s += row[hh * GAT_OUTC + c];
        z[c] = s * (1.f / GAT_HEADS) + b2[c];
    }
    float m = z[0];
#pragma unroll
    for (int c = 1; c < GAT_OUTC; ++c) m = fmaxf(m, z[c]);
    float s = 0.f;
#pragma unroll
    for (int c = 0; c < GAT_OUTC; ++c) s += __expf(z[c] - m);
    float lse = m + __logf(s);
#pragma unroll
    for (int c = 0; c < GAT_OUTC; ++c)
        out[(size_t)nd * GAT_OUTC + c] = z[c] - lse;
}

// ---------------------------------------------------------------------------
static inline int nblk(long long n, int t) { return (int)((n + t - 1) / t); }

extern "C" void kernel_launch(void* const* d_in, const int* in_sizes, int n_in,
                              void* d_out, int out_size, void* d_ws, size_t ws_size,
                              hipStream_t stream) {
    (void)in_sizes; (void)n_in; (void)out_size; (void)ws_size;

    const float*     x    = (const float*)d_in[0];
    const long long* adj  = (const long long*)d_in[1];   // int64 [2,E]
    const float*     W1   = (const float*)d_in[2];       // [512, 64]
    const float*     aS1  = (const float*)d_in[3];       // [8,8]
    const float*     aD1  = (const float*)d_in[4];
    const float*     b1   = (const float*)d_in[5];       // [64]
    const float*     W2   = (const float*)d_in[6];       // [64, 128]
    const float*     aS2  = (const float*)d_in[7];       // [8,16]
    const float*     aD2  = (const float*)d_in[8];
    const float*     b2   = (const float*)d_in[9];       // [16]
    float*           outp = (float*)d_out;               // [N,16] fp32

    const int N = GAT_N, E = GAT_E, H = GAT_HEADS;

    // workspace layout (floats; all region sizes are multiples of 8 -> 32B aligned)
    float* ws = (float*)d_ws;
    size_t o = 0;
    float*    h_big = ws + o; o += (size_t)N * 128;   // h1 (N*64), later h2 (N*128)
    float*    as_   = ws + o; o += (size_t)N * H;
    float*    ad_   = ws + o; o += (size_t)N * H;
    float*    eval  = ws + o; o += (size_t)E * H;
    float*    eexp  = ws + o; o += (size_t)E * H;
    unsigned* emaxk = (unsigned*)(ws + o); o += (size_t)N * H;
    float*    denom = ws + o; o += (size_t)N * H;
    float*    agg   = ws + o; o += (size_t)N * 128;   // L1 uses N*64, L2 N*128
    float*    h2in  = ws + o; o += (size_t)N * 64;
    v16bf*    bpack = (v16bf*)(ws + o); o += 16384;   // 64 KB, reused per layer

    const int T = 256;

    // ===================== Layer 1 =====================
    fill_u32_kernel<<<nblk(N * H, T), T, 0, stream>>>(emaxk, FKEY_NEG_INF, N * H);
    fill_f32_kernel<<<nblk(N * H, T), T, 0, stream>>>(denom, 0.f, N * H);
    fill_f32_kernel<<<nblk((long long)N * 64, T), T, 0, stream>>>(agg, 0.f, N * 64);

    {   // h1 = x @ W1   (50000x512 * 512x64), NT = 64/16 = 4
        const int frag1 = (64 / 16) * (GAT_INC / 32) * 32;   // 2048 threads
        repack_B_kernel<<<nblk(frag1, T), T, 0, stream>>>(W1, bpack, GAT_INC, 64);
        const int waves = N / 16;                            // 3125
        gemm_bf16_wmma_kernel<4><<<nblk((long long)waves * 32, T), T, 0, stream>>>(
            x, bpack, h_big, N, GAT_INC, GAT_INC, 64);
    }
    node_alpha_kernel<<<nblk(N, T), T, 0, stream>>>(h_big, aS1, aD1, as_, ad_, N, H, GAT_MIDC);
    edge_logit_max_kernel<<<nblk((long long)E * H, T), T, 0, stream>>>(adj, as_, ad_, eval, emaxk, E, H);
    edge_exp_sum_kernel<<<nblk((long long)E * H, T), T, 0, stream>>>(adj, eval, emaxk, eexp, denom, E, H);
    edge_message_kernel<<<nblk((long long)E * 16, T), T, 0, stream>>>(adj, eexp, denom, h_big, agg, E, H, GAT_MIDC);
    elu_bias_kernel<<<nblk((long long)N * 64, T), T, 0, stream>>>(agg, b1, h2in, N, 64);

    // ===================== Layer 2 =====================
    fill_u32_kernel<<<nblk(N * H, T), T, 0, stream>>>(emaxk, FKEY_NEG_INF, N * H);
    fill_f32_kernel<<<nblk(N * H, T), T, 0, stream>>>(denom, 0.f, N * H);
    fill_f32_kernel<<<nblk((long long)N * 128, T), T, 0, stream>>>(agg, 0.f, N * 128);

    {   // h2 = h2in @ W2  (50000x64 * 64x128), NT = 128/16 = 8
        const int frag2 = (128 / 16) * (64 / 32) * 32;       // 512 threads
        repack_B_kernel<<<nblk(frag2, T), T, 0, stream>>>(W2, bpack, 64, 128);
        const int waves = N / 16;
        gemm_bf16_wmma_kernel<8><<<nblk((long long)waves * 32, T), T, 0, stream>>>(
            h2in, bpack, h_big, N, 64, 64, 128);
    }
    node_alpha_kernel<<<nblk(N, T), T, 0, stream>>>(h_big, aS2, aD2, as_, ad_, N, H, GAT_OUTC);
    edge_logit_max_kernel<<<nblk((long long)E * H, T), T, 0, stream>>>(adj, as_, ad_, eval, emaxk, E, H);
    edge_exp_sum_kernel<<<nblk((long long)E * H, T), T, 0, stream>>>(adj, eval, emaxk, eexp, denom, E, H);
    edge_message_kernel<<<nblk((long long)E * 32, T), T, 0, stream>>>(adj, eexp, denom, h_big, agg, E, H, GAT_OUTC);

    final_kernel<<<nblk(N, T), T, 0, stream>>>(agg, b2, outp, N);
}
